// MambaDecoderLayer_1511828488829
// MI455X (gfx1250) — matrix operand order repaired and checked by reading, not compiled, not run on hardware
//
#include <hip/hip_runtime.h>

// ---------------------------------------------------------------------------
// Types for gfx1250 WMMA
// ---------------------------------------------------------------------------
typedef __attribute__((ext_vector_type(16))) __bf16 bf16x16;
typedef __attribute__((ext_vector_type(8)))  float  f32x8;

// round-to-nearest-even f32 -> bf16, packed pair into one dword
__device__ __forceinline__ unsigned int pack_bf16x2(float lo, float hi) {
    unsigned int a = __float_as_uint(lo);
    unsigned int b = __float_as_uint(hi);
    a = (a + 0x7FFFu + ((a >> 16) & 1u)) >> 16;
    b = (b + 0x7FFFu + ((b >> 16) & 1u)) & 0xFFFF0000u;
    return (a & 0xFFFFu) | b;
}
__device__ __forceinline__ unsigned short f32_to_bf16(float f) {
    unsigned int u = __float_as_uint(f);
    return (unsigned short)((u + 0x7FFFu + ((u >> 16) & 1u)) >> 16);
}

// ---------------------------------------------------------------------------
// bf16 fragment loads (operands pre-converted; inner loop is pure b128+wmma)
// A (16x32, MxK): lane<16 -> row m0+lane, K={0..7,16..23}; lane>=16 -> K={8..15,24..31}
// ---------------------------------------------------------------------------
__device__ __forceinline__ bf16x16 load_a_frag(const __bf16* __restrict__ X,
                                               int ldx, int m0, int kk) {
    const int lane = threadIdx.x & 31;
    const __bf16* p = X + (size_t)(m0 + (lane & 15)) * ldx + kk + ((lane >> 4) << 3);
    union { uint4 q[2]; bf16x16 v; } r;
    r.q[0] = *(const uint4*)p;          // K + {0..7}
    r.q[1] = *(const uint4*)(p + 16);   // K + {16..23}
    __builtin_prefetch(p + 32, 0, 3);   // next K-slice -> global_prefetch_b8
    return r.v;
}

// B (32x16, KxN), W stored [N,K] row-major:
// lane<16 -> col n0+lane, K={0..7,16..23} ; lane>=16 -> K={8..15,24..31}
__device__ __forceinline__ bf16x16 load_b_frag(const __bf16* __restrict__ W,
                                               int ldw, int n0, int kk) {
    const int lane = threadIdx.x & 31;
    const __bf16* p = W + (size_t)(n0 + (lane & 15)) * ldw + kk + ((lane >> 4) << 3);
    union { uint4 q[2]; bf16x16 v; } r;
    r.q[0] = *(const uint4*)p;
    r.q[1] = *(const uint4*)(p + 16);
    __builtin_prefetch(p + 32, 0, 3);
    return r.v;
}

// ---------------------------------------------------------------------------
// Tiled WMMA GEMM:  Y[M,N] = epilogue( X[M,K] * W[N,K]^T ), bf16 in, f32 out
// block = 256 threads = 8 waves (2 M x 4 N), wave tile 32x64, block tile 64x256
// all dims divide exactly for this layer -> no bounds checks
// ---------------------------------------------------------------------------
enum { EPI_NONE = 0, EPI_SOFTPLUS_BIAS = 1, EPI_RESIDUAL = 2 };

template <int EPI>
__global__ void __launch_bounds__(256)
gemm_bf16_wmma(const __bf16* __restrict__ X, int ldx,
               const __bf16* __restrict__ W, int ldw,
               const float* __restrict__ bias,
               const float* __restrict__ res, int ldr,
               float* __restrict__ Y, int ldy, int K) {
    const int wave = threadIdx.x >> 5;
    const int lane = threadIdx.x & 31;
    const int wm = wave >> 2;          // 0..1
    const int wn = wave & 3;           // 0..3
    const int m_base = blockIdx.y * 64 + wm * 32;
    const int n_base = blockIdx.x * 256 + wn * 64;

    f32x8 acc[2][4];
#pragma unroll
    for (int i = 0; i < 2; ++i)
#pragma unroll
        for (int j = 0; j < 4; ++j) acc[i][j] = {};

    for (int kk = 0; kk < K; kk += 32) {
        const bf16x16 a0 = load_a_frag(X, ldx, m_base,      kk);
        const bf16x16 a1 = load_a_frag(X, ldx, m_base + 16, kk);
#pragma unroll
        for (int j = 0; j < 4; ++j) {
            const bf16x16 bj = load_b_frag(W, ldw, n_base + 16 * j, kk);
            acc[0][j] = __builtin_amdgcn_wmma_f32_16x16x32_bf16(
                false, a0, false, bj, (short)0, acc[0][j], false, false);
            acc[1][j] = __builtin_amdgcn_wmma_f32_16x16x32_bf16(
                false, a1, false, bj, (short)0, acc[1][j], false, false);
        }
    }

    // C/D layout: lane<16 -> rows m..m+7, col=lane ; lane>=16 -> rows m+8..m+15
#pragma unroll
    for (int i = 0; i < 2; ++i) {
        const int row0 = m_base + 16 * i + ((lane >> 4) << 3);
#pragma unroll
        for (int j = 0; j < 4; ++j) {
            const int col = n_base + 16 * j + (lane & 15);
#pragma unroll
            for (int r = 0; r < 8; ++r) {
                float v = acc[i][j][r];
                const int row = row0 + r;
                if (EPI == EPI_SOFTPLUS_BIAS) {
                    float t = v + bias[col];
                    v = (t > 20.f) ? t : log1pf(__expf(t));
                } else if (EPI == EPI_RESIDUAL) {
                    v += res[(size_t)row * ldr + col];
                }
                Y[(size_t)row * ldy + col] = v;
            }
        }
    }
}

// ---------------------------------------------------------------------------
// f32 -> bf16 packed converters (once per operand per launch; bandwidth-bound)
// ---------------------------------------------------------------------------
__global__ void __launch_bounds__(256)
cvt_f32_bf16(const float* __restrict__ in, unsigned int* __restrict__ out,
             unsigned int npairs) {
    const unsigned int i = blockIdx.x * 256 + threadIdx.x;
    if (i < npairs) {
        const float2 f = ((const float2*)in)[i];
        out[i] = pack_bf16x2(f.x, f.y);
    }
}

// dt_raw slice [2048 rows, cols 5120..5247 of zxbcdt(ld=5376)] -> bf16 [2048,128]
__global__ void __launch_bounds__(256)
cvt_dtraw_bf16(const float* __restrict__ zxb, unsigned int* __restrict__ out) {
    const unsigned int i = blockIdx.x * 256 + threadIdx.x;   // pair index
    if (i < 2048u * 64u) {
        const unsigned int r = i >> 6, cp = i & 63u;
        const float2 f = ((const float2*)(zxb + (size_t)r * 5376 + 5120))[cp];
        out[i] = pack_bf16x2(f.x, f.y);
    }
}

// ---------------------------------------------------------------------------
// RMSNorm: one block per token; emits bf16 pairs for the following GEMM
// ---------------------------------------------------------------------------
__global__ void __launch_bounds__(256)
rmsnorm_bf16_kernel(const float* __restrict__ x, const float* __restrict__ w,
                    unsigned int* __restrict__ y, int D) {
    const size_t base = (size_t)blockIdx.x * D;
    float ss = 0.f;
    for (int i = threadIdx.x; i < D; i += 256) {
        const float v = x[base + i];
        ss += v * v;
    }
#pragma unroll
    for (int m = 16; m >= 1; m >>= 1) ss += __shfl_xor(ss, m);
    __shared__ float parts[8];
    if ((threadIdx.x & 31) == 0) parts[threadIdx.x >> 5] = ss;
    __syncthreads();
    float tot = 0.f;
#pragma unroll
    for (int i = 0; i < 8; ++i) tot += parts[i];
    const float rs = rsqrtf(tot / (float)D + 1e-5f);
    const int half = D >> 1;
    for (int i = threadIdx.x; i < half; i += 256) {
        const float2 v = ((const float2*)(x + base))[i];
        const float2 g = ((const float2*)w)[i];
        y[(base >> 1) + i] = pack_bf16x2(v.x * rs * g.x, v.y * rs * g.y);
    }
}

// ---------------------------------------------------------------------------
// Selective scan, fused with D-skip + SiLU(z) gate; writes bf16 y_gated.
// grid = B*64 heads; block = 256 threads; thread owns h[p][nq..nq+3] in regs.
// ---------------------------------------------------------------------------
__global__ void __launch_bounds__(256)
scan_kernel(const float* __restrict__ zxb,        // [2048, 5376] z|x|B|C|dt_raw
            const float* __restrict__ dtb,        // [2048, 2048] softplus(dt)
            const float* __restrict__ A_log,      // [2048, 32]
            const float* __restrict__ Dp,         // [2048]
            unsigned short* __restrict__ yg)      // [2048, 2048] bf16 out
{
    const int LDZ = 5376, L = 1024;
    const int b = blockIdx.x >> 6;
    const int g = blockIdx.x & 63;
    const int tid = threadIdx.x;
    const int p  = tid >> 3;
    const int nq = (tid & 7) << 2;
    const int gkv = g >> 2;            // GQA: 4 heads share x/B group
    const int c = g * 32 + p;

    float Areg[4];
#pragma unroll
    for (int j = 0; j < 4; ++j)
        Areg[j] = -__expf(A_log[(size_t)c * 32 + nq + j]);
    const float Dv = Dp[c];

    float h[4] = {0.f, 0.f, 0.f, 0.f};
    const size_t rowz = (size_t)b * 1024 * LDZ;
    const size_t rowd = (size_t)b * 1024 * 2048;

    float dt_c = 0.f, x_c = 0.f, z_c = 0.f, B_c[4] = {}, C_c[4] = {};
    auto LOAD = [&](int l, float& dt_, float& x_, float* B_, float* C_, float& z_) {
        const size_t rz = rowz + (size_t)l * LDZ;
        dt_ = dtb[rowd + (size_t)l * 2048 + c];
        x_  = zxb[rz + 2048 + gkv * 32 + p];
        z_  = zxb[rz + c];
#pragma unroll
        for (int j = 0; j < 4; ++j) {
            B_[j] = zxb[rz + 2560 + gkv * 32 + nq + j];
            C_[j] = zxb[rz + 3072 + g * 32 + nq + j];
        }
    };
    LOAD(0, dt_c, x_c, B_c, C_c, z_c);

    for (int l = 0; l < L; ++l) {
        float dt_n = 0.f, x_n = 0.f, z_n = 0.f, B_n[4] = {}, C_n[4] = {};
        if (l + 1 < L) LOAD(l + 1, dt_n, x_n, B_n, C_n, z_n);  // pipelined

        const float dtx = dt_c * x_c;
        float ys = 0.f;
#pragma unroll
        for (int j = 0; j < 4; ++j) {
            h[j] = __expf(dt_c * Areg[j]) * h[j] + dtx * B_c[j];
            ys += h[j] * C_c[j];
        }
        ys += __shfl_xor(ys, 1);
        ys += __shfl_xor(ys, 2);
        ys += __shfl_xor(ys, 4);
        if ((tid & 7) == 0) {
            const float zz = z_c;
            const float yy = (ys + x_c * Dv) * (zz / (1.f + __expf(-zz)));
            yg[rowd + (size_t)l * 2048 + c] = f32_to_bf16(yy);
        }
        dt_c = dt_n; x_c = x_n; z_c = z_n;
#pragma unroll
        for (int j = 0; j < 4; ++j) { B_c[j] = B_n[j]; C_c[j] = C_n[j]; }
    }
}

// ---------------------------------------------------------------------------
// SwiGLU elementwise: mlp = bf16( up * silu(gate) )
// ---------------------------------------------------------------------------
__global__ void __launch_bounds__(256)
swiglu_kernel(const float* __restrict__ g, const float* __restrict__ u,
              unsigned int* __restrict__ out, unsigned int npairs) {
    const unsigned int i = blockIdx.x * 256 + threadIdx.x;
    if (i < npairs) {
        const float2 gv = ((const float2*)g)[i];
        const float2 uv = ((const float2*)u)[i];
        const float a = uv.x * (gv.x / (1.f + __expf(-gv.x)));
        const float b = uv.y * (gv.y / (1.f + __expf(-gv.y)));
        out[i] = pack_bf16x2(a, b);
    }
}

// ---------------------------------------------------------------------------
// Launch
// ---------------------------------------------------------------------------
extern "C" void kernel_launch(void* const* d_in, const int* in_sizes, int n_in,
                              void* d_out, int out_size, void* d_ws, size_t ws_size,
                              hipStream_t stream) {
    const float* hidden = (const float*)d_in[0];   // [2,1024,2048]
    const float* Win    = (const float*)d_in[1];   // [5376,2048]
    const float* Wdt    = (const float*)d_in[2];   // [2048,128]
    const float* bdt    = (const float*)d_in[3];   // [2048]
    const float* A_log  = (const float*)d_in[4];   // [2048,32]
    const float* Dp     = (const float*)d_in[5];   // [2048]
    const float* Wout   = (const float*)d_in[6];   // [2048,2048]
    const float* Wg     = (const float*)d_in[7];   // [5632,2048]
    const float* Wu     = (const float*)d_in[8];   // [5632,2048]
    const float* Wd     = (const float*)d_in[9];   // [2048,5632]
    const float* n1     = (const float*)d_in[10];  // [2048]
    const float* n2     = (const float*)d_in[11];  // [2048]
    float* out = (float*)d_out;

    const int M = 2048;                 // B*L tokens
    char* ws = (char*)d_ws;
    auto alloc = [&](size_t bytes) {
        void* p = ws;
        ws += (bytes + 255) & ~(size_t)255;
        return p;
    };
    // f32 intermediates
    float* zxb   = (float*)alloc((size_t)M * 5376 * 4);   // z|x|B|C|dt_raw
    float* dtbuf = (float*)alloc((size_t)M * 2048 * 4);   // softplus(dt)
    float* hbuf  = (float*)alloc((size_t)M * 2048 * 4);   // residual h
    float* gbuf  = (float*)alloc((size_t)M * 5632 * 4);   // gate proj
    float* ubuf  = (float*)alloc((size_t)M * 5632 * 4);   // up proj
    // bf16 GEMM operands (as dword-pair / element views)
    unsigned int* xnb_u   = (unsigned int*)alloc((size_t)M * 2048 * 2);   // xn1 / xn2
    unsigned int* dtraw_u = (unsigned int*)alloc((size_t)M * 128 * 2);
    unsigned short* ygb_u = (unsigned short*)alloc((size_t)M * 2048 * 2);
    unsigned int* mlpb_u  = (unsigned int*)alloc((size_t)M * 5632 * 2);
    unsigned int* Winb_u  = (unsigned int*)alloc((size_t)5376 * 2048 * 2);
    unsigned int* Wdtb_u  = (unsigned int*)alloc((size_t)2048 * 128 * 2);
    unsigned int* Woutb_u = (unsigned int*)alloc((size_t)2048 * 2048 * 2);
    unsigned int* Wgb_u   = (unsigned int*)alloc((size_t)5632 * 2048 * 2);
    unsigned int* Wub_u   = (unsigned int*)alloc((size_t)5632 * 2048 * 2);
    unsigned int* Wdb_u   = (unsigned int*)alloc((size_t)2048 * 5632 * 2);

    const __bf16* xnb   = (const __bf16*)xnb_u;
    const __bf16* dtraw = (const __bf16*)dtraw_u;
    const __bf16* ygb   = (const __bf16*)ygb_u;
    const __bf16* mlpb  = (const __bf16*)mlpb_u;
    const __bf16* Winb  = (const __bf16*)Winb_u;
    const __bf16* Wdtb  = (const __bf16*)Wdtb_u;
    const __bf16* Woutb = (const __bf16*)Woutb_u;
    const __bf16* Wgb   = (const __bf16*)Wgb_u;
    const __bf16* Wub   = (const __bf16*)Wub_u;
    const __bf16* Wdb   = (const __bf16*)Wdb_u;

    auto cvt = [&](const float* src, unsigned int* dst, size_t elems) {
        const unsigned int np = (unsigned int)(elems / 2);
        cvt_f32_bf16<<<(np + 255) / 256, 256, 0, stream>>>(src, dst, np);
    };

    // 0) weights -> bf16 (once per launch; ~15us of HBM traffic total)
    cvt(Win,  Winb_u,  (size_t)5376 * 2048);
    cvt(Wdt,  Wdtb_u,  (size_t)2048 * 128);
    cvt(Wout, Woutb_u, (size_t)2048 * 2048);
    cvt(Wg,   Wgb_u,   (size_t)5632 * 2048);
    cvt(Wu,   Wub_u,   (size_t)5632 * 2048);
    cvt(Wd,   Wdb_u,   (size_t)2048 * 5632);

    // 1) xn1 = rmsnorm(hidden, norm1) -> bf16
    rmsnorm_bf16_kernel<<<M, 256, 0, stream>>>(hidden, n1, xnb_u, 2048);

    // 2) zxbcdt = xn1 @ in_proj_w.T            [2048,5376] f32
    gemm_bf16_wmma<EPI_NONE><<<dim3(5376 / 256, M / 64), 256, 0, stream>>>(
        xnb, 2048, Winb, 2048, nullptr, nullptr, 0, zxb, 5376, 2048);

    // 3) dt_raw slice -> bf16 ; dt = softplus(dt_raw @ dt_proj_w.T + b)
    cvt_dtraw_bf16<<<(2048 * 64 + 255) / 256, 256, 0, stream>>>(zxb, dtraw_u);
    gemm_bf16_wmma<EPI_SOFTPLUS_BIAS><<<dim3(2048 / 256, M / 64), 256, 0, stream>>>(
        dtraw, 128, Wdtb, 128, bdt, nullptr, 0, dtbuf, 2048, 128);

    // 4) selective scan + D-skip + silu(z) gate -> bf16 y_gated
    scan_kernel<<<128, 256, 0, stream>>>(zxb, dtbuf, A_log, Dp, ygb_u);

    // 5) h = hidden + y_gated @ out_proj_w.T   f32
    gemm_bf16_wmma<EPI_RESIDUAL><<<dim3(2048 / 256, M / 64), 256, 0, stream>>>(
        ygb, 2048, Woutb, 2048, nullptr, hidden, 2048, hbuf, 2048, 2048);

    // 6) xn2 = rmsnorm(h, norm2) -> bf16 (reuses xn buffer)
    rmsnorm_bf16_kernel<<<M, 256, 0, stream>>>(hbuf, n2, xnb_u, 2048);

    // 7) gate / up projections      [2048,5632] f32
    gemm_bf16_wmma<EPI_NONE><<<dim3(5632 / 256, M / 64), 256, 0, stream>>>(
        xnb, 2048, Wgb, 2048, nullptr, nullptr, 0, gbuf, 5632, 2048);
    gemm_bf16_wmma<EPI_NONE><<<dim3(5632 / 256, M / 64), 256, 0, stream>>>(
        xnb, 2048, Wub, 2048, nullptr, nullptr, 0, ubuf, 5632, 2048);

    // 8) mlp = bf16( up * silu(gate) )
    const unsigned int nmlp2 = (unsigned int)((size_t)M * 5632 / 2);
    swiglu_kernel<<<(nmlp2 + 255) / 256, 256, 0, stream>>>(gbuf, ubuf, mlpb_u, nmlp2);

    // 9) out = h + mlp @ down_w.T   K=5632
    gemm_bf16_wmma<EPI_RESIDUAL><<<dim3(2048 / 256, M / 64), 256, 0, stream>>>(
        mlpb, 5632, Wdb, 5632, nullptr, hbuf, 2048, out, 2048, 5632);
}
